// ThreeLayerFCModel_53927609369198
// MI455X (gfx1250) — matrix operand order repaired and checked by reading, not compiled, not run on hardware
//
#include <hip/hip_runtime.h>

// ---------------------------------------------------------------------------
// Fused 3-layer MLP for MI455X (gfx1250, wave32, WMMA bf16)
//   layer: y = relu(x @ W^T + b), all dims 1024, B = 16384 rows
// WMMA role swap: A := W tile (16n x 32k), B := act tile (32k x 16m),
// D := H^T tile. D lanes hold 8 consecutive output columns of one activation
// row -> b128 epilogue stores; bias preloaded into accumulators.
// ---------------------------------------------------------------------------

typedef __attribute__((ext_vector_type(16))) __bf16 v16bf;
typedef __attribute__((ext_vector_type(8)))  float  v8f;

#define DIM     1024
#define TM      64            // rows per workgroup
#define STRIDE  1032          // bf16 elements per LDS row (+16B pad: bank-conflict-free)
#define SMEM_BYTES (2 * TM * STRIDE * 2)

__device__ __forceinline__ unsigned short f32_to_bf16(float f) {
  unsigned int u = __float_as_uint(f);
  u += 0x7FFFu + ((u >> 16) & 1u);          // round to nearest even
  return (unsigned short)(u >> 16);
}

// pack two f32 -> two bf16 in one dword: single v_perm_b32 (high halves)
__device__ __forceinline__ unsigned int pack_bf16(float a, float b) {
  return __builtin_amdgcn_perm(__float_as_uint(b), __float_as_uint(a),
                               0x07060302u);
}

// single-instruction ReLU: v_med3_f32(x, 0, +inf)
__device__ __forceinline__ float relu(float x) {
#if __has_builtin(__builtin_amdgcn_fmed3f)
  return __builtin_amdgcn_fmed3f(x, 0.0f, __builtin_huge_valf());
#else
  return fmaxf(x, 0.0f);
#endif
}

union Frag {                                 // one 16x32 / 32x16 bf16 fragment
  uint4 q[2];                                // 32 bytes per lane
  v16bf v;
};

// ---- prep: w[n,k] * mask[n,k] -> bf16, row-major [N,K] ---------------------
__global__ void cvt_weights(const float* __restrict__ w,
                            const unsigned char* __restrict__ mask,
                            unsigned short* __restrict__ out, int n) {
  int i = blockIdx.x * blockDim.x + threadIdx.x;
  int stride = gridDim.x * blockDim.x;
  for (; i < n; i += stride) {
    float v = w[i];
    if (mask != nullptr && mask[i] == 0) v = 0.0f;
    out[i] = f32_to_bf16(v);
  }
}

// ---- fused 3-layer MLP -----------------------------------------------------
__global__ __launch_bounds__(256, 1) void fused_mlp(
    const float* __restrict__ x,
    const unsigned short* __restrict__ wb0,
    const unsigned short* __restrict__ wb1,
    const unsigned short* __restrict__ wb2,
    const float* __restrict__ bias0,
    const float* __restrict__ bias1,
    const float* __restrict__ bias2,
    float* __restrict__ out) {
  extern __shared__ unsigned short smem[];
  unsigned short* actA = smem;
  unsigned short* actB = smem + TM * STRIDE;

  const int tid  = threadIdx.x;
  const int lane = tid & 31;
  const int wv   = tid >> 5;     // wave 0..7
  const int lLo  = lane & 15;
  const int lHi  = lane >> 4;    // 0 or 1
  const int row0 = blockIdx.x * TM;

  // ---- stage x tile: f32 global -> bf16 LDS (actA), row-major ----
  for (int idx = tid; idx < TM * (DIM / 4); idx += 256) {
    int r  = idx >> 8;           // 256 float4 per row
    int c4 = idx & 255;
    float4 f = ((const float4*)(x + (size_t)(row0 + r) * DIM))[c4];
    *(uint2*)&actA[r * STRIDE + c4 * 4] =
        make_uint2(pack_bf16(f.x, f.y), pack_bf16(f.z, f.w));
  }

  const unsigned short* wl[3] = {wb0, wb1, wb2};
  const float*          bl[3] = {bias0, bias1, bias2};

#pragma unroll
  for (int layer = 0; layer < 3; ++layer) {
    const unsigned short* __restrict__ W    = wl[layer];
    const float*          __restrict__ bias = bl[layer];
    unsigned short* ain  = (layer & 1) ? actB : actA;
    unsigned short* aout = (layer & 1) ? actA : actB;
    __syncthreads();   // ain fully produced (x stage or previous layer)

    // wave wv owns output columns [wv*128, wv*128+128): 4 pairs of 16-wide tiles
#pragma unroll 1
    for (int p = 0; p < 4; ++p) {
      const int n0 = wv * 128 + p * 32;

      // preload bias (8 consecutive n per lane, uniform over lLo) -> acc init
      v8f binit[2];
#pragma unroll
      for (int j = 0; j < 2; ++j) {
        const float* bp = bias + n0 + j * 16 + lHi * 8;
        float4 lo = *(const float4*)bp;
        float4 hi = *((const float4*)bp + 1);
        binit[j] = (v8f){lo.x, lo.y, lo.z, lo.w, hi.x, hi.y, hi.z, hi.w};
      }
      v8f acc[2][4];
#pragma unroll
      for (int j = 0; j < 2; ++j)
#pragma unroll
        for (int m = 0; m < 4; ++m) acc[j][m] = binit[j];

#pragma unroll 1
      for (int k0 = 0; k0 < DIM; k0 += 32) {
        // A fragments = weights: lane -> W row n0+j*16+lLo,
        // K chunks at k0+lHi*8 (+0..7) and +16 (A-matrix interleave)
        Frag afr[2];
#pragma unroll
        for (int j = 0; j < 2; ++j) {
          const unsigned short* wp =
              W + (size_t)(n0 + j * 16 + lLo) * DIM + k0 + lHi * 8;
          afr[j].q[0] = *(const uint4*)(wp);
          afr[j].q[1] = *(const uint4*)(wp + 16);
        }
        // B fragments = activations: lane -> act row m*16+lLo,
        // K chunks at k0+lHi*8 (+0..7) and +16 (same interleave as A)
#pragma unroll
        for (int m = 0; m < 4; ++m) {
          Frag b;
          const unsigned short* ap =
              ain + (m * 16 + lLo) * STRIDE + k0 + lHi * 8;
          b.q[0] = *(const uint4*)(ap);
          b.q[1] = *(const uint4*)(ap + 16);
          acc[0][m] = __builtin_amdgcn_wmma_f32_16x16x32_bf16(
              false, afr[0].v, false, b.v, (short)0, acc[0][m], false, false);
          acc[1][m] = __builtin_amdgcn_wmma_f32_16x16x32_bf16(
              false, afr[1].v, false, b.v, (short)0, acc[1][m], false, false);
        }
      }

      // epilogue: relu (bias already in acc).
      // D layout: lane = act row (m*16+lLo), VGPR v = out col n0+j*16+lHi*8+v
#pragma unroll
      for (int j = 0; j < 2; ++j) {
#pragma unroll
        for (int m = 0; m < 4; ++m) {
          v8f r;
#pragma unroll
          for (int v = 0; v < 8; ++v) r[v] = relu(acc[j][m][v]);
          if (layer == 2) {
            float* op = out + (size_t)(row0 + m * 16 + lLo) * DIM +
                        n0 + j * 16 + lHi * 8;
            *(float4*)op       = make_float4(r[0], r[1], r[2], r[3]);
            *((float4*)op + 1) = make_float4(r[4], r[5], r[6], r[7]);
          } else {
            uint4 pk;
            pk.x = pack_bf16(r[0], r[1]);
            pk.y = pack_bf16(r[2], r[3]);
            pk.z = pack_bf16(r[4], r[5]);
            pk.w = pack_bf16(r[6], r[7]);
            *(uint4*)&aout[(m * 16 + lLo) * STRIDE + n0 + j * 16 + lHi * 8] = pk;
          }
        }
      }
    }
  }
}

// ---------------------------------------------------------------------------
extern "C" void kernel_launch(void* const* d_in, const int* in_sizes, int n_in,
                              void* d_out, int out_size, void* d_ws, size_t ws_size,
                              hipStream_t stream) {
  (void)in_sizes; (void)n_in; (void)out_size; (void)ws_size;
  // setup_inputs order: x, w0, b0, w1, b1, mask1, w2, b2, mask2
  const float*         x  = (const float*)d_in[0];
  const float*         w0 = (const float*)d_in[1];
  const float*         b0 = (const float*)d_in[2];
  const float*         w1 = (const float*)d_in[3];
  const float*         b1 = (const float*)d_in[4];
  const unsigned char* m1 = (const unsigned char*)d_in[5];
  const float*         w2 = (const float*)d_in[6];
  const float*         b2 = (const float*)d_in[7];
  const unsigned char* m2 = (const unsigned char*)d_in[8];

  unsigned short* wb0 = (unsigned short*)d_ws;                     // 2 MB each
  unsigned short* wb1 = wb0 + (size_t)DIM * DIM;
  unsigned short* wb2 = wb1 + (size_t)DIM * DIM;
  const int nW = DIM * DIM;

  cvt_weights<<<512, 256, 0, stream>>>(w0, nullptr, wb0, nW);
  cvt_weights<<<512, 256, 0, stream>>>(w1, m1,      wb1, nW);
  cvt_weights<<<512, 256, 0, stream>>>(w2, m2,      wb2, nW);

  hipFuncSetAttribute((const void*)fused_mlp,
                      hipFuncAttributeMaxDynamicSharedMemorySize, SMEM_BYTES);
  fused_mlp<<<16384 / TM, 256, SMEM_BYTES, stream>>>(
      x, wb0, wb1, wb2, b0, b1, b2, (float*)d_out);
}
